// MSAAttention_85581518340271
// MI455X (gfx1250) — compile-verified, matrix operand order checked by reading, not run on hardware
//
#include <hip/hip_runtime.h>
#include <hip/hip_bf16.h>

typedef __bf16 bf16;
typedef __attribute__((ext_vector_type(16))) __bf16 v16bf;
typedef __attribute__((ext_vector_type(8)))  float  v8f;
typedef __attribute__((ext_vector_type(4))) unsigned int u32x4;
typedef __attribute__((ext_vector_type(8))) int i32x8;
typedef __attribute__((ext_vector_type(4))) int i32x4;

#ifndef USE_TDM
#if __has_builtin(__builtin_amdgcn_tensor_load_to_lds) && __has_builtin(__builtin_amdgcn_s_wait_tensorcnt)
#define USE_TDM 1
#else
#define USE_TDM 0
#endif
#endif

// ---------------------------------------------------------------------------
// TDM: async 2D tile load global->LDS (bf16 elements, optional row padding).
// D# per CDNA5 ISA. Issued by one wave; completion via
// s_wait_tensorcnt. pad codes: interval 2^(c+1) DWORDs, amount (c+1) DWORDs.
// This toolchain exposes the 6-arg builtin: (g0, g1, g2, g3, g2b, cpol).
// ---------------------------------------------------------------------------
#if USE_TDM
__device__ inline void tdm_load_2d_bf16(const void* gptr, void* ldsptr,
                                        unsigned rows, unsigned row_elems,
                                        unsigned src_stride_elems,
                                        unsigned pad_interval_code,
                                        unsigned pad_amount_code) {
    unsigned long long ga = (unsigned long long)(uintptr_t)gptr;
    unsigned lds = (unsigned)(uintptr_t)ldsptr;   // low 32 bits = LDS byte offset
    u32x4 g0;
    g0[0] = 1u;                                        // count=1 (valid)
    g0[1] = lds;                                       // lds_addr
    g0[2] = (unsigned)(ga & 0xFFFFFFFFu);              // global_addr[31:0]
    g0[3] = (unsigned)((ga >> 32) & 0x1FFFFFFu) | (2u << 30);  // [56:32] | type=2
    i32x8 g1;
    g1[0] = (int)((1u << 16)                           // data_size = 2 bytes
                | (1u << 20)                           // pad_enable
                | (pad_interval_code << 22)
                | (pad_amount_code << 25));
    g1[1] = (int)(row_elems << 16);                    // tensor_dim0[15:0]
    g1[2] = (int)(rows << 16);                         // tensor_dim1[15:0]
    g1[3] = (int)(row_elems << 16);                    // tile_dim0
    g1[4] = (int)rows;                                 // tile_dim1 (tile_dim2=0)
    g1[5] = (int)src_stride_elems;                     // tensor_dim0_stride[31:0]
    g1[6] = 0;
    g1[7] = 0;
    i32x4 z4 = {0, 0, 0, 0};
    i32x8 z8 = {0, 0, 0, 0, 0, 0, 0, 0};
    __builtin_amdgcn_tensor_load_to_lds(g0, g1, z4, z4, z8, 0);
}
#endif

// ---------------------------------------------------------------------------
// Fragment gather from an LDS tile laid out [idx][k] (contraction-contiguous).
// 16-bit operand layout (ISA 7.12.2): lane L holds idx = idx0 + (L&15),
// half = L>>4; element e maps to K = k0 + e + 8*half + (e>=8 ? 8 : 0).
// ---------------------------------------------------------------------------
__device__ inline v16bf load_frag(const bf16* base, int idx0, int k0, int ldk) {
    int lane = threadIdx.x & 31;
    int half = lane >> 4;
    const bf16* p = base + (idx0 + (lane & 15)) * ldk + k0 + 8 * half;
    v16bf f;
#pragma unroll
    for (int e = 0; e < 16; ++e) {
        int kk = (e < 8) ? e : (e + 8);
        f[e] = p[kk];
    }
    return f;
}

// ---------------------------------------------------------------------------
__global__ void cvt_f32_bf16(const float* __restrict__ in, bf16* __restrict__ out, int n) {
    int i = blockIdx.x * 256 + threadIdx.x;
    if (i < n) out[i] = (bf16)in[i];
}

// ---------------------------------------------------------------------------
// Generic bf16 WMMA GEMM:  C[M,N] = A[M,K] * W[K,N] + bias
// Block tile 128x128, K-step 32. 8 waves in 2(M) x 4(N); 8 v_wmma/wave/iter.
// A tile staged by TDM (padded row stride), B tile staged transposed by VALU.
// ---------------------------------------------------------------------------
template <bool OUT_BF16>
__global__ __launch_bounds__(256)
void gemm_bf16_wmma(const bf16* __restrict__ A, int lda,
                    const bf16* __restrict__ W, int ldw,
                    const float* __restrict__ bias,
                    void* __restrict__ Cout, int ldc, int K) {
    constexpr int BM = 128, BN = 128, BK = 32;
    constexpr int LDA_S = BK + 8;   // 40 bf16 (TDM pad: 16DW interval + 4DW pad)
    constexpr int LDB_S = BK + 8;   // 40, B stored [n][k]
    __shared__ bf16 sA[BM * LDA_S];
    __shared__ bf16 sB[BN * LDB_S];

    const int tid = threadIdx.x;
    const int wave = tid >> 5;
    const int m0 = blockIdx.x * BM;
    const int n0 = blockIdx.y * BN;
    const int wm = (wave >> 2) * 64;
    const int wn = (wave & 3) * 32;

    v8f acc[4][2] = {};

    for (int k0 = 0; k0 < K; k0 += BK) {
        __builtin_prefetch(W + (size_t)(k0 + BK) * ldw + n0, 0, 1);
#if USE_TDM
        if (wave == 0)
            tdm_load_2d_bf16(A + (size_t)m0 * lda + k0, sA, BM, BK, (unsigned)lda, 3, 3);
#else
        for (int i = tid; i < BM * BK / 2; i += 256) {
            int e = i * 2;
            int r = e / BK, c = e % BK;
            uint32_t v = *reinterpret_cast<const uint32_t*>(A + (size_t)(m0 + r) * lda + k0 + c);
            *reinterpret_cast<uint32_t*>(&sA[r * LDA_S + c]) = v;
        }
#endif
        // Stage B tile transposed: global [k][n] -> LDS [n][k]
        for (int i = tid; i < BK * BN / 2; i += 256) {
            int e = i * 2;
            int kk = e / BN, n = e % BN;
            uint32_t v = *reinterpret_cast<const uint32_t*>(W + (size_t)(k0 + kk) * ldw + n0 + n);
            bf16 b0 = ((const bf16*)&v)[0];
            bf16 b1 = ((const bf16*)&v)[1];
            sB[n * LDB_S + kk]       = b0;
            sB[(n + 1) * LDB_S + kk] = b1;
        }
#if USE_TDM
        if (wave == 0) __builtin_amdgcn_s_wait_tensorcnt(0);
#endif
        __syncthreads();

        v16bf af[4], bf[2];
#pragma unroll
        for (int i = 0; i < 4; ++i) af[i] = load_frag(sA, wm + 16 * i, 0, LDA_S);
#pragma unroll
        for (int j = 0; j < 2; ++j) bf[j] = load_frag(sB, wn + 16 * j, 0, LDB_S);
#pragma unroll
        for (int i = 0; i < 4; ++i)
#pragma unroll
            for (int j = 0; j < 2; ++j)
                acc[i][j] = __builtin_amdgcn_wmma_f32_16x16x32_bf16(
                    false, af[i], false, bf[j], (short)0, acc[i][j], false, false);
        __syncthreads();
    }

    int lane  = tid & 31;
    int nlane = lane & 15;
    int mhalf = (lane >> 4) * 8;
#pragma unroll
    for (int i = 0; i < 4; ++i)
#pragma unroll
        for (int j = 0; j < 2; ++j) {
            int gn = n0 + wn + 16 * j + nlane;
            float bv = bias[gn];
#pragma unroll
            for (int r = 0; r < 8; ++r) {
                int gm = m0 + wm + 16 * i + mhalf + r;
                float v = acc[i][j][r] + bv;
                if (OUT_BF16) ((bf16*)Cout)[(size_t)gm * ldc + gn] = (bf16)v;
                else          ((float*)Cout)[(size_t)gm * ldc + gn] = v;
            }
        }
}

// ---------------------------------------------------------------------------
// Flash attention, 8 heads, d=64. qkv: [B,N,1536] bf16 (q|k|v channels).
// 256 threads (8 waves), 128 q-rows/block, 64-key tiles.
// SCALE folded into Q fragments (power of two -> exact in bf16).
// ---------------------------------------------------------------------------
__global__ __launch_bounds__(256)
void attn_wmma(const bf16* __restrict__ qkv, bf16* __restrict__ ctx) {
    constexpr int QB = 128, KB = 64, D = 64;
    constexpr int LDK = D + 8;    // sK [key][d]  (TDM pad: 32DW interval + 4DW)
    constexpr int LDV = KB + 8;   // sV [d][key]  (V transposed)
    constexpr int LDP = KB + 8;   // sP per-wave [row][key]

    __shared__ bf16 sK[KB * LDK];
    __shared__ bf16 sV[D * LDV];
    __shared__ bf16 sP[8 * 16 * LDP];   // also reused as Q staging (128*72)

    const int tid  = threadIdx.x;
    const int wave = tid >> 5;
    const int lane = tid & 31;
    const int nlane = lane & 15;
    const int mhalf = (lane >> 4) * 8;

    const int b = blockIdx.y >> 3;
    const int h = blockIdx.y & 7;
    const int q0 = blockIdx.x * QB;
    const bf16* qbase = qkv + (size_t)b * 2048 * 1536 + h * 64;
    const bf16* kbase = qbase + 512;
    const bf16* vbase = qbase + 1024;

    // ---- Stage Q block into sP region, build persistent A-fragments ----
    bf16* sQ = sP;
#if USE_TDM
    if (wave == 0) {
        tdm_load_2d_bf16(qbase + (size_t)q0 * 1536, sQ, QB, D, 1536, 4, 3);
        __builtin_amdgcn_s_wait_tensorcnt(0);
    }
#else
    for (int i = tid; i < QB * D / 2; i += 256) {
        int e = i * 2;
        int r = e / D, d = e % D;
        uint32_t v = *reinterpret_cast<const uint32_t*>(qbase + (size_t)(q0 + r) * 1536 + d);
        *reinterpret_cast<uint32_t*>(&sQ[r * LDK + d]) = v;
    }
#endif
    __syncthreads();
    v16bf qf[2];
#pragma unroll
    for (int ks = 0; ks < 2; ++ks) qf[ks] = load_frag(sQ, wave * 16, ks * 32, LDK);
    // fold softmax scale (64^-0.5 = 0.125, exact power of two) into Q
#pragma unroll
    for (int ks = 0; ks < 2; ++ks)
#pragma unroll
        for (int e = 0; e < 16; ++e) qf[ks][e] = (bf16)((float)qf[ks][e] * 0.125f);
    __syncthreads();   // before sP reuse

    float rowm[8], rowl[8];
#pragma unroll
    for (int r = 0; r < 8; ++r) { rowm[r] = -3.0e38f; rowl[r] = 0.0f; }
    v8f o[4] = {};

    for (int j = 0; j < 2048 / KB; ++j) {
#if USE_TDM
        if (wave == 0)
            tdm_load_2d_bf16(kbase + (size_t)(j * KB) * 1536, sK, KB, D, 1536, 4, 3);
        for (int i = tid; i < KB * D / 2; i += 256) {
            int e = i * 2;
            int key = e / D, d = e % D;
            uint32_t vv = *reinterpret_cast<const uint32_t*>(vbase + (size_t)(j * KB + key) * 1536 + d);
            sV[d * LDV + key]       = ((const bf16*)&vv)[0];
            sV[(d + 1) * LDV + key] = ((const bf16*)&vv)[1];
        }
        if (wave == 0) __builtin_amdgcn_s_wait_tensorcnt(0);
#else
        for (int i = tid; i < KB * D / 2; i += 256) {
            int e = i * 2;
            int key = e / D, d = e % D;
            uint32_t kv = *reinterpret_cast<const uint32_t*>(kbase + (size_t)(j * KB + key) * 1536 + d);
            *reinterpret_cast<uint32_t*>(&sK[key * LDK + d]) = kv;
            uint32_t vv = *reinterpret_cast<const uint32_t*>(vbase + (size_t)(j * KB + key) * 1536 + d);
            sV[d * LDV + key]       = ((const bf16*)&vv)[0];
            sV[(d + 1) * LDV + key] = ((const bf16*)&vv)[1];
        }
#endif
        __syncthreads();

        // ---- S = (Q*scale) K^T ----
        v8f s[4];
#pragma unroll
        for (int kt = 0; kt < 4; ++kt) {
            v8f z = {};
#pragma unroll
            for (int ks = 0; ks < 2; ++ks) {
                v16bf kf = load_frag(sK, kt * 16, ks * 32, LDK);
                z = __builtin_amdgcn_wmma_f32_16x16x32_bf16(
                    false, qf[ks], false, kf, (short)0, z, false, false);
            }
            s[kt] = z;
        }

        // ---- online softmax (row r lives at element r across 16 lanes) ----
        float mnew[8], corr[8], psum[8];
#pragma unroll
        for (int r = 0; r < 8; ++r) {
            float t = s[0][r];
#pragma unroll
            for (int kt = 1; kt < 4; ++kt) t = fmaxf(t, s[kt][r]);
#pragma unroll
            for (int m = 1; m < 16; m <<= 1) t = fmaxf(t, __shfl_xor(t, m, 32));
            mnew[r] = fmaxf(rowm[r], t);
            corr[r] = __expf(rowm[r] - mnew[r]);
            psum[r] = 0.0f;
        }
#pragma unroll
        for (int kt = 0; kt < 4; ++kt)
#pragma unroll
            for (int r = 0; r < 8; ++r) {
                float pv = __expf(s[kt][r] - mnew[r]);
                psum[r] += pv;
                sP[(wave * 16 + mhalf + r) * LDP + kt * 16 + nlane] = (bf16)pv;
            }
#pragma unroll
        for (int r = 0; r < 8; ++r) {
            float t = psum[r];
#pragma unroll
            for (int m = 1; m < 16; m <<= 1) t += __shfl_xor(t, m, 32);
            rowl[r] = rowl[r] * corr[r] + t;
            rowm[r] = mnew[r];
        }
#pragma unroll
        for (int dt = 0; dt < 4; ++dt)
#pragma unroll
            for (int r = 0; r < 8; ++r) o[dt][r] *= corr[r];

        // ---- O += P V ----
        v16bf pf[2];
#pragma unroll
        for (int ks = 0; ks < 2; ++ks) pf[ks] = load_frag(sP, wave * 16, ks * 32, LDP);
#pragma unroll
        for (int dt = 0; dt < 4; ++dt)
#pragma unroll
            for (int ks = 0; ks < 2; ++ks) {
                v16bf vf = load_frag(sV, dt * 16, ks * 32, LDV);
                o[dt] = __builtin_amdgcn_wmma_f32_16x16x32_bf16(
                    false, pf[ks], false, vf, (short)0, o[dt], false, false);
            }
        __syncthreads();
    }

    // ---- normalize and write ctx[b,n,h*64+d] ----
#pragma unroll
    for (int dt = 0; dt < 4; ++dt)
#pragma unroll
        for (int r = 0; r < 8; ++r) {
            int n = q0 + wave * 16 + mhalf + r;
            int d = dt * 16 + nlane;
            float v = o[dt][r] / rowl[r];
            ctx[((size_t)b * 2048 + n) * 512 + h * 64 + d] = (bf16)v;
        }
}

// ---------------------------------------------------------------------------
// IBPP branch: per token-row, [16] @ w_in + b_in, SiLU, @ w_out + b_out.
// ---------------------------------------------------------------------------
__global__ __launch_bounds__(256)
void ibpp_kernel(const float* __restrict__ x,
                 const float* __restrict__ w_in, const float* __restrict__ b_in,
                 const float* __restrict__ w_out, const float* __restrict__ b_out,
                 bf16* __restrict__ cc) {
    int gid = blockIdx.x * 256 + threadIdx.x;
    int token = gid >> 4;
    int row = gid & 15;
    const float* xr = x + (size_t)token * 256 + row * 16;

    float t[16];
#pragma unroll
    for (int j = 0; j < 16; ++j) t[j] = b_in[j];
#pragma unroll
    for (int k = 0; k < 16; ++k) {
        float xv = xr[k];
#pragma unroll
        for (int j = 0; j < 16; ++j) t[j] = fmaf(xv, w_in[k * 16 + j], t[j]);
    }
#pragma unroll
    for (int j = 0; j < 16; ++j) t[j] = t[j] / (1.0f + __expf(-t[j]));

    float o[16];
#pragma unroll
    for (int j = 0; j < 16; ++j) o[j] = b_out[j];
#pragma unroll
    for (int k = 0; k < 16; ++k) {
        float tv = t[k];
#pragma unroll
        for (int j = 0; j < 16; ++j) o[j] = fmaf(tv, w_out[k * 16 + j], o[j]);
    }
    bf16* dst = cc + (size_t)token * 768 + 512 + row * 16;
#pragma unroll
    for (int j = 0; j < 16; ++j) dst[j] = (bf16)o[j];
}

// ---------------------------------------------------------------------------
extern "C" void kernel_launch(void* const* d_in, const int* in_sizes, int n_in,
                              void* d_out, int out_size, void* d_ws, size_t ws_size,
                              hipStream_t stream) {
    (void)in_sizes; (void)n_in; (void)out_size; (void)ws_size;
    const float* x       = (const float*)d_in[0];
    const float* w_qkv   = (const float*)d_in[1];
    const float* b_qkv   = (const float*)d_in[2];
    const float* w_in    = (const float*)d_in[3];
    const float* b_in    = (const float*)d_in[4];
    const float* w_out   = (const float*)d_in[5];
    const float* b_out   = (const float*)d_in[6];
    const float* w_merge = (const float*)d_in[7];
    const float* b_merge = (const float*)d_in[8];
    const float* w_proj  = (const float*)d_in[9];
    const float* b_proj  = (const float*)d_in[10];
    float* out = (float*)d_out;

    char* ws = (char*)d_ws;
    bf16* xb      = (bf16*)(ws);                       // 8192*256
    bf16* wqkvb   = (bf16*)(ws + 4194304);             // 256*1536
    bf16* wmergeb = (bf16*)(ws + 4980736);             // 512*512
    bf16* wprojb  = (bf16*)(ws + 5505024);             // 768*512
    bf16* qkvb    = (bf16*)(ws + 6291456);             // 8192*1536
    bf16* ctx     = (bf16*)(ws + 31457280);            // 8192*512
    bf16* cc      = (bf16*)(ws + 39845888);            // 8192*768

    cvt_f32_bf16<<<8192, 256, 0, stream>>>(x, xb, 8192 * 256);
    cvt_f32_bf16<<<1536, 256, 0, stream>>>(w_qkv, wqkvb, 256 * 1536);
    cvt_f32_bf16<<<1024, 256, 0, stream>>>(w_merge, wmergeb, 512 * 512);
    cvt_f32_bf16<<<1536, 256, 0, stream>>>(w_proj, wprojb, 768 * 512);

    gemm_bf16_wmma<true><<<dim3(64, 12), 256, 0, stream>>>(
        xb, 256, wqkvb, 1536, b_qkv, qkvb, 1536, 256);

    attn_wmma<<<dim3(16, 32), 256, 0, stream>>>(qkvb, ctx);

    gemm_bf16_wmma<true><<<dim3(64, 4), 256, 0, stream>>>(
        ctx, 512, wmergeb, 512, b_merge, cc, 768, 512);

    ibpp_kernel<<<512, 256, 0, stream>>>(x, w_in, b_in, w_out, b_out, cc);

    gemm_bf16_wmma<false><<<dim3(64, 4), 256, 0, stream>>>(
        cc, 768, wprojb, 512, b_proj, out, 512, 768);
}
